// RGLRU_48103633715283
// MI455X (gfx1250) — compile-verified, matrix operand order checked by reading, not run on hardware
//
#include <hip/hip_runtime.h>

// ---------------------------------------------------------------------------
// RG-LRU (Griffin) for MI455X / gfx1250.
//   Kernel 1: decompose block-diag gate weights into bf16 hi/lo WMMA B-frags.
//   Kernel 2: per-(64-token, head) WG: GEMM gates via v_wmma_f32_16x16x32_bf16
//             (bf16x3 for ~f32 accuracy) + fused elementwise epilogue,
//             emitting a_eff and x_norm to workspace.
//   Kernel 3: bandwidth-bound linear scan, one thread per (b,w) column.
// Workspace requirement: 4 MB frags + 2 * B*L*W*4 = ~260 MB.
// ---------------------------------------------------------------------------

#define DEV_INLINE __device__ __forceinline__

typedef __attribute__((ext_vector_type(16))) __bf16 v16bf;
typedef __attribute__((ext_vector_type(8)))  float  v8f;

constexpr int B_  = 4;
constexpr int L_  = 4096;
constexpr int W_  = 2048;
constexpr int H_  = 8;
constexpr int BW_ = 256;          // block width per head

constexpr int MC      = 64;       // tokens per workgroup (4 WMMA M-tiles)
constexpr int XPITCH  = 260;      // LDS row pitch (floats), 16B-aligned rows
constexpr int NTILES  = BW_ / 16; // 16 N-tiles per head
constexpr int KSTEPS  = BW_ / 32; // 8 K-steps of 32 for bf16 WMMA
constexpr int NFRAG   = 2 * H_ * NTILES * KSTEPS;   // 2048 B-fragments

// 16-bit A/B matrix VGPR layout (CDNA5 ISA §7.12.2):
// element e (0..15) of v16bf, lane-half -> K index within a 32-wide K step.
DEV_INLINE int kmap16(int e, int half) {
  const int v = e >> 1;       // packed dword index 0..7
  const int p = e & 1;
  const int base = (v < 4) ? 0 : 16;
  return base + 2 * (v & 3) + p + 8 * half;
}

// ---------------------------------------------------------------------------
// Kernel 1: weights -> bf16 hi/lo fragments in WMMA-B per-lane layout.
// One thread per (fragment, lane).  frag id f = ((g*H + h)*16 + nt)*8 + ks.
// ---------------------------------------------------------------------------
__global__ __launch_bounds__(256) void rglru_prep_w(
    const float* __restrict__ ig_w, const float* __restrict__ ag_w,
    v16bf* __restrict__ wHi, v16bf* __restrict__ wLo) {
  const int gid  = blockIdx.x * blockDim.x + threadIdx.x;
  const int lane = gid & 31;
  const int f    = gid >> 5;
  if (f >= NFRAG) return;
  const int ks = f & 7;
  const int nt = (f >> 3) & 15;
  const int h  = (f >> 7) & 7;
  const int g  = f >> 10;
  const float* src = (g == 0 ? ig_w : ag_w) + (size_t)h * BW_ * BW_;
  const int n    = nt * 16 + (lane & 15);
  const int half = lane >> 4;
  v16bf hi, lo;
#pragma unroll
  for (int e = 0; e < 16; ++e) {
    const int k = ks * 32 + kmap16(e, half);
    const float v = src[(size_t)k * BW_ + n];   // w[h][i=k][j=n]
    const __bf16 hh = (__bf16)v;
    hi[e] = hh;
    lo[e] = (__bf16)(v - (float)hh);
  }
  wHi[(size_t)f * 32 + lane] = hi;
  wLo[(size_t)f * 32 + lane] = lo;
}

// ---------------------------------------------------------------------------
// Kernel 2: gate GEMMs (bf16x3 WMMA) + elementwise epilogue.
// Grid: (B*L/MC) * H workgroups of 256 threads (8 waves).
// Wave w owns N-tiles {2w, 2w+1} for both gates across 4 M-tiles.
// ---------------------------------------------------------------------------
__global__ __launch_bounds__(256) void rglru_gates(
    const float* __restrict__ x, const int* __restrict__ segment_pos,
    const float* __restrict__ ig_b, const float* __restrict__ ag_b,
    const float* __restrict__ a_param,
    const v16bf* __restrict__ wHi, const v16bf* __restrict__ wLo,
    float* __restrict__ wsA, float* __restrict__ wsX) {
  __shared__ float ldsX[MC * XPITCH];
  __shared__ int   ldsSeg[MC];

  const int tid    = threadIdx.x;
  const int lane   = tid & 31;
  const int wave   = tid >> 5;
  const int h      = blockIdx.x & (H_ - 1);
  const int chunk  = blockIdx.x >> 3;       // over flattened B*L / MC
  const int token0 = chunk * MC;            // flat token index b*L + l
  const int l0     = token0 & (L_ - 1);     // MC divides L, so chunk stays in one b

  // ---- stage x chunk [MC x BW] into LDS (float4, coalesced) ----
  {
    const int j4 = (tid & 63) << 2;         // 0..252
    const int m0 = tid >> 6;                // 0..3
    for (int m = m0; m < MC; m += 4) {
      const float4 v = *reinterpret_cast<const float4*>(
          &x[(size_t)(token0 + m) * W_ + h * BW_ + j4]);
      *reinterpret_cast<float4*>(&ldsX[m * XPITCH + j4]) = v;
    }
    if (tid < MC) ldsSeg[tid] = segment_pos[l0 + tid];
  }
  __syncthreads();

  v8f acc[2][2][4];                          // [gate][ntile][Mtile]
#pragma unroll
  for (int g = 0; g < 2; ++g)
#pragma unroll
    for (int ti = 0; ti < 2; ++ti)
#pragma unroll
      for (int mt = 0; mt < 4; ++mt)
        acc[g][ti][mt] = v8f{0.f, 0.f, 0.f, 0.f, 0.f, 0.f, 0.f, 0.f};

  const int half  = lane >> 4;
  const int mlane = lane & 15;

  for (int ks = 0; ks < KSTEPS; ++ks) {
    // build A hi/lo fragments for the 4 M-tiles from LDS x
    v16bf ahi[4], alo[4];
#pragma unroll
    for (int mt = 0; mt < 4; ++mt) {
#pragma unroll
      for (int e = 0; e < 16; ++e) {
        const int k = ks * 32 + kmap16(e, half);
        const float xv = ldsX[(mt * 16 + mlane) * XPITCH + k];
        const __bf16 hh = (__bf16)xv;
        ahi[mt][e] = hh;
        alo[mt][e] = (__bf16)(xv - (float)hh);
      }
    }
#pragma unroll
    for (int g = 0; g < 2; ++g) {
#pragma unroll
      for (int ti = 0; ti < 2; ++ti) {
        const int nt = wave * 2 + ti;
        const size_t f = ((((size_t)g * H_ + h) * NTILES + nt) * KSTEPS + ks);
        const v16bf bhi = wHi[f * 32 + lane];
        const v16bf blo = wLo[f * 32 + lane];
#pragma unroll
        for (int mt = 0; mt < 4; ++mt) {
          // bf16x3: hi*hi + hi*lo + lo*hi ~= full f32 product
          acc[g][ti][mt] = __builtin_amdgcn_wmma_f32_16x16x32_bf16(
              false, ahi[mt], false, bhi, (short)0, acc[g][ti][mt], false, false);
          acc[g][ti][mt] = __builtin_amdgcn_wmma_f32_16x16x32_bf16(
              false, ahi[mt], false, blo, (short)0, acc[g][ti][mt], false, false);
          acc[g][ti][mt] = __builtin_amdgcn_wmma_f32_16x16x32_bf16(
              false, alo[mt], false, bhi, (short)0, acc[g][ti][mt], false, false);
        }
      }
    }
  }

  // ---- elementwise epilogue straight from C-matrix registers ----
  // C layout: element r -> row M = r + 8*(lane>=16), col N = lane%16.
#pragma unroll
  for (int ti = 0; ti < 2; ++ti) {
    const int j    = (wave * 2 + ti) * 16 + mlane;  // column within head
    const int wcol = h * BW_ + j;                   // global channel
    const float ap = a_param[wcol];
    const float sp = (ap > 20.f) ? ap : log1pf(expf(ap));  // softplus
    const float bi = ig_b[wcol];
    const float ba = ag_b[wcol];
#pragma unroll
    for (int mt = 0; mt < 4; ++mt) {
#pragma unroll
      for (int r = 0; r < 8; ++r) {
        const int mrow = mt * 16 + r + 8 * half;
        const float xv = ldsX[mrow * XPITCH + j];
        const float gi = 1.f / (1.f + expf(-(acc[0][ti][mt][r] + bi)));
        const float ga = 1.f / (1.f + expf(-(acc[1][ti][mt][r] + ba)));
        const float log_a = -8.f * ga * sp;
        const float a  = expf(log_a);
        const float a2 = expf(2.f * log_a);
        const bool reset = (ldsSeg[mrow] == 0);
        const float mult = reset ? 1.f : sqrtf(fmaxf(1.f - a2, 0.f));
        const size_t off = (size_t)(token0 + mrow) * W_ + wcol;
        wsA[off] = reset ? 0.f : a;
        wsX[off] = xv * gi * mult;
      }
    }
  }
}

// ---------------------------------------------------------------------------
// Kernel 3: linear scan h_t = a_t*h + x_t. One thread per (b,w) column;
// lane dimension = consecutive w -> fully coalesced loads/stores.
// ---------------------------------------------------------------------------
__global__ __launch_bounds__(256) void rglru_scan(
    const float* __restrict__ wsA, const float* __restrict__ wsX,
    const float* __restrict__ prev_h,
    float* __restrict__ y, float* __restrict__ last_h) {
  const int idx = blockIdx.x * blockDim.x + threadIdx.x;  // b*W + w
  const int b = idx / W_;
  const int w = idx - b * W_;
  float hst = prev_h[idx];
  size_t o = (size_t)b * L_ * W_ + w;
#pragma unroll 8
  for (int t = 0; t < L_; ++t, o += W_) {
    hst = fmaf(wsA[o], hst, wsX[o]);
    y[o] = hst;
  }
  last_h[idx] = hst;
}

// ---------------------------------------------------------------------------
extern "C" void kernel_launch(void* const* d_in, const int* in_sizes, int n_in,
                              void* d_out, int out_size, void* d_ws, size_t ws_size,
                              hipStream_t stream) {
  (void)in_sizes; (void)n_in; (void)out_size; (void)ws_size;

  const float* x       = (const float*)d_in[0];
  const int*   seg     = (const int*)d_in[1];
  const float* prev_h  = (const float*)d_in[2];
  const float* ig_w    = (const float*)d_in[3];
  const float* ig_b    = (const float*)d_in[4];
  const float* ag_w    = (const float*)d_in[5];
  const float* ag_b    = (const float*)d_in[6];
  const float* a_param = (const float*)d_in[7];

  char* ws = (char*)d_ws;
  const size_t FRAG_BYTES = (size_t)NFRAG * 32 * sizeof(v16bf);  // 2 MB
  v16bf* wHi = (v16bf*)(ws);
  v16bf* wLo = (v16bf*)(ws + FRAG_BYTES);
  float* wsA = (float*)(ws + 2 * FRAG_BYTES);
  float* wsX = wsA + (size_t)B_ * L_ * W_;

  float* y      = (float*)d_out;
  float* last_h = y + (size_t)B_ * L_ * W_;

  rglru_prep_w<<<(NFRAG * 32) / 256, 256, 0, stream>>>(ig_w, ag_w, wHi, wLo);
  rglru_gates<<<(B_ * L_ / MC) * H_, 256, 0, stream>>>(
      x, seg, ig_b, ag_b, a_param, wHi, wLo, wsA, wsX);
  rglru_scan<<<(B_ * W_) / 256, 256, 0, stream>>>(wsA, wsX, prev_h, y, last_h);
}
